// DeformConv_30760555774159
// MI455X (gfx1250) — matrix operand order repaired and checked by reading, not compiled, not run on hardware
//
#include <hip/hip_runtime.h>
#include <hip/hip_bf16.h>
#include <math.h>

typedef __attribute__((ext_vector_type(16))) __bf16 v16bf;
typedef __attribute__((ext_vector_type(8)))  __bf16 v8bf;
typedef __attribute__((ext_vector_type(8)))  float  v8f;
typedef __attribute__((ext_vector_type(4)))  int    v4i;

#define BN   4
#define CINC 256
#define HH   64
#define WW   64

#if defined(__has_builtin)
#if __has_builtin(__builtin_amdgcn_global_load_async_to_lds_b128)
#define HAVE_ASYNC_LDS 1
#endif
#endif

#if defined(HAVE_ASYNC_LDS)
typedef __attribute__((address_space(1))) v4i GV4;
typedef __attribute__((address_space(3))) v4i LV4;
#define ASYNC_CP_B128(gsrc, ldst) \
    __builtin_amdgcn_global_load_async_to_lds_b128((GV4*)(gsrc), (LV4*)(ldst), 0, 0)
#define ASYNC_WAIT() asm volatile("s_wait_asynccnt 0x0" ::: "memory")
#else
#define ASYNC_WAIT()
#endif

__device__ __forceinline__ float sigmoidf_(float v) { return 1.f / (1.f + __expf(-v)); }

// ---- pack conv weights f32 [OC][CIN][3][3] -> bf16 [9][OCpad][CIN] (zero-padded rows)
__global__ void pack_w_kernel(const float* __restrict__ w, __bf16* __restrict__ wp,
                              int OC, int OCpad) {
    int i = blockIdx.x * blockDim.x + threadIdx.x;
    int total = 9 * OCpad * CINC;
    if (i >= total) return;
    int tap = i / (OCpad * CINC);
    int r   = i - tap * (OCpad * CINC);
    int o   = r / CINC;
    int c   = r - o * CINC;
    float v = 0.f;
    if (o < OC) v = w[(o * CINC + c) * 9 + tap];
    wp[i] = (__bf16)v;
}

// ---- NCHW f32 -> NHWC bf16  ([b][y][x][c])
__global__ void nhwc_kernel(const float* __restrict__ x, __bf16* __restrict__ xt) {
    int i = blockIdx.x * blockDim.x + threadIdx.x;
    int total = BN * CINC * HH * WW;
    if (i >= total) return;
    int xc = i % WW;
    int y  = (i / WW) % HH;
    int c  = (i / (WW * HH)) % CINC;
    int b  = i / (WW * HH * CINC);
    xt[(((size_t)b * HH + y) * WW + xc) * CINC + c] = (__bf16)x[i];
}

// ---- implicit-GEMM 3x3 conv (pad=1) on bf16 NHWC input
// 64 cout x 64 pixels (one image row) per block; bf16 WMMA, f32 accumulate.
// outBF != nullptr  -> SiLU + bf16 NHWC output (stride OC)
// outF  != nullptr  -> plain bias + f32 NHWC output (stride OC)
__global__ void __launch_bounds__(256)
conv3x3_wmma_kernel(const __bf16* __restrict__ in, __bf16* __restrict__ outBF,
                    float* __restrict__ outF,
                    const __bf16* __restrict__ wp, const float* __restrict__ bias,
                    int OC, int OCpad) {
    __shared__ __align__(32) __bf16 lds[3 * 66 * 32];   // [row][col][32 ch]
    const int tid    = threadIdx.x;
    const int lane   = tid & 31;
    const int wave   = tid >> 5;
    const int waveM  = wave & 3;
    const int waveN  = wave >> 2;
    const int coutBase = blockIdx.x * 64 + waveM * 16;
    const int b = blockIdx.y / HH;
    const int y = blockIdx.y % HH;
    const int laneM  = lane & 15;
    const int laneHi = lane >> 4;

    v8f acc0 = {}; v8f acc1 = {};

    for (int cc = 0; cc < 8; ++cc) {          // cin chunks of 32
        const int cin0 = cc * 32;
        __syncthreads();
        // stage [3][66][32ch] bf16 halo patch: 792 x 16B transfers
        for (int i = tid; i < 198 * 4; i += 256) {
            int pos = i >> 2;                 // 0..197
            int q   = i & 3;                  // 8-channel quad
            int r   = pos / 66;
            int c   = pos - r * 66;
            int gy = y - 1 + r;
            int gx = c - 1;
            __bf16* dst = &lds[(r * 66 + c) * 32 + q * 8];
            if (gy >= 0 && gy < HH && gx >= 0 && gx < WW) {
                const __bf16* src =
                    in + (((size_t)b * HH + gy) * WW + gx) * CINC + cin0 + q * 8;
#if defined(HAVE_ASYNC_LDS)
                ASYNC_CP_B128(src, dst);
#else
                *reinterpret_cast<v8bf*>(dst) = *reinterpret_cast<const v8bf*>(src);
#endif
            } else {
                v8bf z = {};
                *reinterpret_cast<v8bf*>(dst) = z;
            }
        }
        ASYNC_WAIT();
        __syncthreads();
        for (int tap = 0; tap < 9; ++tap) {
            const int ki = tap / 3, kj = tap % 3;
            // A fragment from packed weights [tap][cout][cin]
            const __bf16* ap = wp + ((size_t)tap * OCpad + coutBase + laneM) * CINC
                                  + cin0 + laneHi * 16;
            union { v16bf v; v8bf h[2]; } a;
            a.h[0] = *reinterpret_cast<const v8bf*>(ap);
            a.h[1] = *reinterpret_cast<const v8bf*>(ap + 8);
            {
                int px = waveN * 32 + laneM;
                const __bf16* bp = &lds[(ki * 66 + (px + kj)) * 32 + laneHi * 16];
                union { v16bf v; v8bf h[2]; } bb;
                bb.h[0] = *reinterpret_cast<const v8bf*>(bp);
                bb.h[1] = *reinterpret_cast<const v8bf*>(bp + 8);
                acc0 = __builtin_amdgcn_wmma_f32_16x16x32_bf16(
                    false, a.v, false, bb.v, (short)0, acc0, false, false);
            }
            {
                int px = waveN * 32 + 16 + laneM;
                const __bf16* bp = &lds[(ki * 66 + (px + kj)) * 32 + laneHi * 16];
                union { v16bf v; v8bf h[2]; } bb;
                bb.h[0] = *reinterpret_cast<const v8bf*>(bp);
                bb.h[1] = *reinterpret_cast<const v8bf*>(bp + 8);
                acc1 = __builtin_amdgcn_wmma_f32_16x16x32_bf16(
                    false, a.v, false, bb.v, (short)0, acc1, false, false);
            }
        }
    }
    // epilogue: NHWC stores; 8 accumulator rows per lane == 8 consecutive channels
    const int row0 = coutBase + laneHi * 8;
    for (int sub = 0; sub < 2; ++sub) {
        v8f acc = sub ? acc1 : acc0;
        int px = waveN * 32 + sub * 16 + laneM;
        size_t pixbase = (((size_t)b * HH + y) * WW + px) * (size_t)OC;
        if (row0 + 7 < OC) {
            if (outBF) {
                v8bf pk;
                #pragma unroll
                for (int j = 0; j < 8; ++j) {
                    float v = acc[j] + bias[row0 + j];
                    v = v * sigmoidf_(v);                 // SiLU
                    pk[j] = (__bf16)v;
                }
                *reinterpret_cast<v8bf*>(&outBF[pixbase + row0]) = pk;   // 16B store
            } else {
                float4 s0, s1;
                s0.x = acc[0] + bias[row0 + 0]; s0.y = acc[1] + bias[row0 + 1];
                s0.z = acc[2] + bias[row0 + 2]; s0.w = acc[3] + bias[row0 + 3];
                s1.x = acc[4] + bias[row0 + 4]; s1.y = acc[5] + bias[row0 + 5];
                s1.z = acc[6] + bias[row0 + 6]; s1.w = acc[7] + bias[row0 + 7];
                *reinterpret_cast<float4*>(&outF[pixbase + row0])     = s0;
                *reinterpret_cast<float4*>(&outF[pixbase + row0 + 4]) = s1;
            }
        }
    }
}

// ---- modulated deformable conv: bilinear sample into LDS, then bf16 WMMA GEMM
// xt: bf16 NHWC;  off3: f32 NHWC (stride 432);  output f32 NCHW (reference layout)
__global__ void __launch_bounds__(256)
deform_wmma_kernel(const __bf16* __restrict__ xt, const float* __restrict__ off3,
                   const __bf16* __restrict__ wp, const float* __restrict__ bias,
                   float* __restrict__ out) {
    __shared__ __align__(32) __bf16 lds[64 * 32];   // [pixel][32 ch] (one group pair)
    const int tid    = threadIdx.x;
    const int lane   = tid & 31;
    const int wave   = tid >> 5;
    const int waveM  = wave & 3;
    const int waveN  = wave >> 2;
    const int coutBase = blockIdx.x * 64 + waveM * 16;
    const int b = blockIdx.y / HH;
    const int y = blockIdx.y % HH;
    const int laneM  = lane & 15;
    const int laneHi = lane >> 4;

    // sampling mapping: 64 px x 2 groups x 2 halves = 256 threads
    const int sp_px   = tid >> 2;
    const int sp_gl   = (tid >> 1) & 1;
    const int sp_half = tid & 1;

    v8f acc0 = {}; v8f acc1 = {};

    for (int g2 = 0; g2 < 8; ++g2) {          // group pairs -> 32-channel K chunks
        for (int tap = 0; tap < 9; ++tap) {
            __syncthreads();
            // ---- cooperative bilinear sampling into LDS
            {
                const int g  = g2 * 2 + sp_gl;
                const int ki = tap / 3, kj = tap % 3;
                const size_t pbase = (((size_t)b * HH + y) * WW + sp_px) * 432;
                float dy = off3[pbase + (18 * g + 2 * tap)];
                float dx = off3[pbase + (18 * g + 2 * tap + 1)];
                float mk = sigmoidf_(off3[pbase + (288 + 9 * g + tap)]);
                float ys = (float)(y - 1 + ki) + dy;
                float xs = (float)(sp_px - 1 + kj) + dx;
                float y0 = floorf(ys), x0 = floorf(xs);
                float wy = ys - y0,   wx = xs - x0;
                float vout[8];
                #pragma unroll
                for (int e = 0; e < 8; ++e) vout[e] = 0.f;
                const int chb = g * 16 + sp_half * 8;
                #pragma unroll
                for (int corner = 0; corner < 4; ++corner) {
                    float yi = y0 + (float)(corner >> 1);
                    float xi = x0 + (float)(corner & 1);
                    float wgt = ((corner >> 1) ? wy : 1.f - wy) *
                                ((corner & 1)  ? wx : 1.f - wx);
                    if (yi >= 0.f && yi < (float)HH && xi >= 0.f && xi < (float)WW) {
                        int yc = (int)yi, xc = (int)xi;
                        const __bf16* src = xt + (((size_t)b * HH + yc) * WW + xc) * CINC + chb;
                        v8bf cv = *reinterpret_cast<const v8bf*>(src);
                        #pragma unroll
                        for (int e = 0; e < 8; ++e) vout[e] += wgt * (float)cv[e];
                    }
                }
                __bf16* dst = &lds[sp_px * 32 + sp_gl * 16 + sp_half * 8];
                #pragma unroll
                for (int e = 0; e < 8; ++e) dst[e] = (__bf16)(vout[e] * mk);
            }
            __syncthreads();
            // ---- WMMA mainloop step
            const __bf16* ap = wp + ((size_t)tap * 256 + coutBase + laneM) * CINC
                                  + g2 * 32 + laneHi * 16;
            union { v16bf v; v8bf h[2]; } a;
            a.h[0] = *reinterpret_cast<const v8bf*>(ap);
            a.h[1] = *reinterpret_cast<const v8bf*>(ap + 8);
            {
                int px = waveN * 32 + laneM;
                const __bf16* bp = &lds[px * 32 + laneHi * 16];
                union { v16bf v; v8bf h[2]; } bb;
                bb.h[0] = *reinterpret_cast<const v8bf*>(bp);
                bb.h[1] = *reinterpret_cast<const v8bf*>(bp + 8);
                acc0 = __builtin_amdgcn_wmma_f32_16x16x32_bf16(
                    false, a.v, false, bb.v, (short)0, acc0, false, false);
            }
            {
                int px = waveN * 32 + 16 + laneM;
                const __bf16* bp = &lds[px * 32 + laneHi * 16];
                union { v16bf v; v8bf h[2]; } bb;
                bb.h[0] = *reinterpret_cast<const v8bf*>(bp);
                bb.h[1] = *reinterpret_cast<const v8bf*>(bp + 8);
                acc1 = __builtin_amdgcn_wmma_f32_16x16x32_bf16(
                    false, a.v, false, bb.v, (short)0, acc1, false, false);
            }
        }
    }
    // output: f32 NCHW (reference output layout)
    for (int sub = 0; sub < 2; ++sub) {
        v8f acc = sub ? acc1 : acc0;
        int px = waveN * 32 + sub * 16 + laneM;
        #pragma unroll
        for (int j = 0; j < 8; ++j) {
            int row = coutBase + laneHi * 8 + j;
            float v = acc[j] + bias[row];
            out[(((size_t)b * 256 + row) * HH + y) * WW + px] = v;
        }
    }
}

extern "C" void kernel_launch(void* const* d_in, const int* in_sizes, int n_in,
                              void* d_out, int out_size, void* d_ws, size_t ws_size,
                              hipStream_t stream) {
    const float* x    = (const float*)d_in[0];
    const float* offf = (const float*)d_in[1];
    const float* wdef = (const float*)d_in[2];
    const float* bdef = (const float*)d_in[3];
    const float* w1   = (const float*)d_in[4];
    const float* b1   = (const float*)d_in[5];
    const float* w2   = (const float*)d_in[6];
    const float* b2   = (const float*)d_in[7];
    const float* w3   = (const float*)d_in[8];
    const float* b3   = (const float*)d_in[9];
    float* out = (float*)d_out;

    char* ws = (char*)d_ws;
    size_t off = 0;
    auto alloc = [&](size_t bytes) {
        void* p = ws + off;
        off += (bytes + 255) & ~(size_t)255;
        return p;
    };
    __bf16* xoff = (__bf16*)alloc((size_t)BN * HH * WW * 256 * 2);  // offset_feat NHWC bf16
    __bf16* t1   = (__bf16*)alloc((size_t)BN * HH * WW * 256 * 2);  // NHWC bf16
    __bf16* t2   = (__bf16*)alloc((size_t)BN * HH * WW * 256 * 2);  // NHWC bf16
    float*  o3   = (float*) alloc((size_t)BN * HH * WW * 432 * 4);  // NHWC f32
    __bf16* w1p  = (__bf16*)alloc((size_t)9 * 256 * 256 * 2);
    __bf16* w2p  = (__bf16*)alloc((size_t)9 * 256 * 256 * 2);
    __bf16* w3p  = (__bf16*)alloc((size_t)9 * 448 * 256 * 2);
    __bf16* wdp  = (__bf16*)alloc((size_t)9 * 256 * 256 * 2);
    __bf16* xt   = (__bf16*)alloc((size_t)BN * HH * WW * 256 * 2);  // x NHWC bf16

    pack_w_kernel<<<(9 * 256 * 256 + 255) / 256, 256, 0, stream>>>(w1, w1p, 256, 256);
    pack_w_kernel<<<(9 * 256 * 256 + 255) / 256, 256, 0, stream>>>(w2, w2p, 256, 256);
    pack_w_kernel<<<(9 * 448 * 256 + 255) / 256, 256, 0, stream>>>(w3, w3p, 432, 448);
    pack_w_kernel<<<(9 * 256 * 256 + 255) / 256, 256, 0, stream>>>(wdef, wdp, 256, 256);
    nhwc_kernel<<<(BN * 256 * HH * WW + 255) / 256, 256, 0, stream>>>(x, xt);
    nhwc_kernel<<<(BN * 256 * HH * WW + 255) / 256, 256, 0, stream>>>(offf, xoff);

    dim3 gconv(4, BN * HH);
    conv3x3_wmma_kernel<<<gconv, 256, 0, stream>>>(xoff, t1, nullptr, w1p, b1, 256, 256);
    conv3x3_wmma_kernel<<<gconv, 256, 0, stream>>>(t1,   t2, nullptr, w2p, b2, 256, 256);
    dim3 gconv3(7, BN * HH);
    conv3x3_wmma_kernel<<<gconv3, 256, 0, stream>>>(t2, nullptr, o3, w3p, b3, 432, 448);

    deform_wmma_kernel<<<dim3(4, BN * HH), 256, 0, stream>>>(xt, o3, wdp, bdef, out);
}
